// CausalVisionMamba_57269093924968
// MI455X (gfx1250) — compile-verified, matrix operand order checked by reading, not run on hardware
//
#include <hip/hip_runtime.h>

// ---------------------------------------------------------------------------
// CausalVisionMamba forward for MI455X (gfx1250, wave32, WMMA bf16,
// async global->LDS staging on interior tiles)
// ---------------------------------------------------------------------------
#define B_SZ 4
#define C_IN 3
#define HW 224
#define D_MODEL 384
#define D_INNER 768
#define D_STATE 16
#define DT_RANK 24
#define D_CONV 4
#define N_LAYERS 4
#define NUM_CLASSES 1000
#define GRID_HW 56
#define SEQ_L (GRID_HW * GRID_HW)   /* 3136  */
#define M_ROWS (B_SZ * SEQ_L)       /* 12544 */
#define DBL_W (DT_RANK + 2 * D_STATE) /* 56 */

typedef __attribute__((ext_vector_type(16))) __bf16 v16bf;
typedef __attribute__((ext_vector_type(8)))  float  v8f;

struct Frag32B { uint4 lo, hi; };   // 32 bytes == one 16-element bf16 fragment

__device__ __forceinline__ unsigned short f32_to_bf16(float f) {
  unsigned int u = __builtin_bit_cast(unsigned int, f);
  u += 0x7FFFu + ((u >> 16) & 1u);           // round-to-nearest-even
  return (unsigned short)(u >> 16);
}
__device__ __forceinline__ float sigmoid_f(float x) { return 1.f / (1.f + __expf(-x)); }

// ---------------------------------------------------------------------------
// Generic elementwise f32 -> bf16 conversion
// ---------------------------------------------------------------------------
__global__ void f2bf_kernel(const float* __restrict__ s, unsigned short* __restrict__ d, int n) {
  int i = blockIdx.x * blockDim.x + threadIdx.x;
  if (i < n) d[i] = f32_to_bf16(s[i]);
}

// ---------------------------------------------------------------------------
// Patch embedding: 4x4 stride-4 conv  ->  t[M_ROWS, D_MODEL] (+ bf16 copy)
// ---------------------------------------------------------------------------
__global__ void patch_embed_kernel(const float* __restrict__ x, const float* __restrict__ pw,
                                   const float* __restrict__ pb, float* __restrict__ t,
                                   unsigned short* __restrict__ tbf) {
  int idx = blockIdx.x * blockDim.x + threadIdx.x;
  if (idx >= M_ROWS * D_MODEL) return;
  int d   = idx % D_MODEL;
  int row = idx / D_MODEL;              // b*SEQ_L + l
  int b   = row / SEQ_L;
  int l   = row % SEQ_L;
  int oh = l / GRID_HW, ow = l % GRID_HW;
  float s = pb[d];
#pragma unroll
  for (int c = 0; c < C_IN; ++c)
#pragma unroll
    for (int kh = 0; kh < 4; ++kh)
#pragma unroll
      for (int kw = 0; kw < 4; ++kw)
        s += x[((size_t)(b * C_IN + c) * HW + (oh * 4 + kh)) * HW + (ow * 4 + kw)] *
             pw[((d * C_IN + c) * 4 + kh) * 4 + kw];
  t[idx]   = s;
  tbf[idx] = f32_to_bf16(s);
}

// ---------------------------------------------------------------------------
// WMMA bf16 GEMM:  C[M,N] = act( A[M,K] * W[N,K]^T + bias ) (+ beta*Cold)
//   block tile 128x64x32, 8 waves (wave32), each wave owns a 32x32 quadrant
//   (4x v_wmma_f32_16x16x32_bf16 per K-step).
//   Interior tiles: double-buffered LDS staged by GLOBAL_LOAD_ASYNC_TO_LDS_B128
//   (ASYNCcnt DMA, overlaps next tile's copy with this tile's WMMAs).
//   Edge tiles: register-pipelined global_load_b128 with clamped addresses and
//   data cndmask zero-fill (EXEC stays all-ones at every WMMA, ISA 7.12).
//   Requires: lda/ldw/K multiples of 8 (true for every call in this model).
// ---------------------------------------------------------------------------
#define BM 128
#define BN 64
#define BK 32
#define APAD 40   // padded LDS row (ushorts): 80B rows -> conflict-free b128 reads

__global__ __launch_bounds__(256)
void gemm_bf16_kernel(const unsigned short* __restrict__ A,
                      const unsigned short* __restrict__ W,
                      const float* __restrict__ bias,
                      float* __restrict__ C,
                      unsigned short* __restrict__ Cbf,
                      int M, int N, int K, int lda, int ldw, int ldc,
                      int act /*0 none, 1 softplus*/, int beta /*0/1 add old C*/) {
  __shared__ __align__(16) unsigned short As[2][BM * APAD];
  __shared__ __align__(16) unsigned short Bs[2][BN * APAD];

  const int tid  = threadIdx.x;
  const int lane = tid & 31;
  const int wave = tid >> 5;
  const int wm   = wave & 3;   // 4 waves along M (32 rows each)
  const int wn   = wave >> 2;  // 2 waves along N (32 cols each)
  const int row0 = blockIdx.y * BM;
  const int col0 = blockIdx.x * BN;

  const int half = lane >> 4;  // 16-lane half (ISA 16-bit A/B striping)
  const int l16  = lane & 15;
  const int klo  = half * 8;

  // Per-thread staging chunk coordinates (16B = 8 bf16 per chunk)
  // A tile: 512 chunks -> 2 per thread;  B tile: 256 chunks -> 1 per thread.
  const int ar0 = (tid) >> 2,        ac0 = (tid & 3) * 8;
  const int ar1 = (tid + 256) >> 2,  ac1 = ac0;
  const int bn0 = tid >> 2,          bc0 = (tid & 3) * 8;

  v8f acc[2][2];
#pragma unroll
  for (int i = 0; i < 2; ++i)
#pragma unroll
    for (int j = 0; j < 2; ++j)
#pragma unroll
      for (int r = 0; r < 8; ++r) acc[i][j][r] = 0.f;

  // Fragment readers (documented 16-bit 16x32 striping:
  //   lane half h holds K {h*8..h*8+7, h*8+16..h*8+23} of row/col l16)
  auto fragA = [&](int s, int i) -> v16bf {
    const unsigned short* rp = &As[s][(wm * 32 + i * 16 + l16) * APAD + klo];
    Frag32B f;
    f.lo = *reinterpret_cast<const uint4*>(rp);
    f.hi = *reinterpret_cast<const uint4*>(rp + 16);
    return __builtin_bit_cast(v16bf, f);
  };
  auto fragB = [&](int s, int j) -> v16bf {
    const unsigned short* rp = &Bs[s][(wn * 32 + j * 16 + l16) * APAD + klo];
    Frag32B f;
    f.lo = *reinterpret_cast<const uint4*>(rp);
    f.hi = *reinterpret_cast<const uint4*>(rp + 16);
    return __builtin_bit_cast(v16bf, f);
  };
  auto do_wmma = [&](int s) {
    v16bf a0 = fragA(s, 0), a1 = fragA(s, 1);
    v16bf b0 = fragB(s, 0), b1 = fragB(s, 1);
    acc[0][0] = __builtin_amdgcn_wmma_f32_16x16x32_bf16(false, a0, false, b0, (short)0, acc[0][0], false, false);
    acc[0][1] = __builtin_amdgcn_wmma_f32_16x16x32_bf16(false, a0, false, b1, (short)0, acc[0][1], false, false);
    acc[1][0] = __builtin_amdgcn_wmma_f32_16x16x32_bf16(false, a1, false, b0, (short)0, acc[1][0], false, false);
    acc[1][1] = __builtin_amdgcn_wmma_f32_16x16x32_bf16(false, a1, false, b1, (short)0, acc[1][1], false, false);
  };

  const bool interior = (row0 + BM <= M) && (col0 + BN <= N) && ((K & (BK - 1)) == 0);

  if (interior) {
    // ---- Fast path: async DMA staging (global_load_async_to_lds_b128) ----
    auto stage_async = [&](int k0, int s) {
      const unsigned short* ga0 = &A[(size_t)(row0 + ar0) * lda + k0 + ac0];
      const unsigned short* ga1 = &A[(size_t)(row0 + ar1) * lda + k0 + ac1];
      const unsigned short* gb  = &W[(size_t)(col0 + bn0) * ldw + k0 + bc0];
      unsigned la0 = (unsigned)(uintptr_t)&As[s][ar0 * APAD + ac0];
      unsigned la1 = (unsigned)(uintptr_t)&As[s][ar1 * APAD + ac1];
      unsigned lb  = (unsigned)(uintptr_t)&Bs[s][bn0 * APAD + bc0];
      asm volatile("global_load_async_to_lds_b128 %0, %1, off" :: "v"(la0), "v"(ga0) : "memory");
      asm volatile("global_load_async_to_lds_b128 %0, %1, off" :: "v"(la1), "v"(ga1) : "memory");
      asm volatile("global_load_async_to_lds_b128 %0, %1, off" :: "v"(lb),  "v"(gb)  : "memory");
    };

    stage_async(0, 0);
    int s = 0;
    for (int k0 = 0; k0 < K; k0 += BK, s ^= 1) {
      asm volatile("s_wait_asynccnt 0x0" ::: "memory");  // own tile copies done
      __syncthreads();                                   // everyone's copies visible
      if (k0 + BK < K) stage_async(k0 + BK, s ^ 1);      // overlap with WMMAs below
      do_wmma(s);
    }
  } else {
    // ---- Edge path: register-pipelined vector loads, data zero-fill ----
    const uint4 z4 = {0u, 0u, 0u, 0u};
    auto fetchA = [&](int k0, int r, int c) -> uint4 {
      int gr = row0 + r, gc = k0 + c;
      bool p = (gr < M) && (gc + 8 <= K);
      int grc = (gr < M) ? gr : (M - 1);
      int gcc = (gc + 8 <= K) ? gc : ((K - 8) & ~7);
      uint4 v = *reinterpret_cast<const uint4*>(&A[(size_t)grc * lda + gcc]);
      return p ? v : z4;
    };
    auto fetchB = [&](int k0, int n, int c) -> uint4 {
      int gn = col0 + n, gc = k0 + c;
      bool p = (gn < N) && (gc + 8 <= K);
      int gnc = (gn < N) ? gn : (N - 1);
      int gcc = (gc + 8 <= K) ? gc : ((K - 8) & ~7);
      uint4 v = *reinterpret_cast<const uint4*>(&W[(size_t)gnc * ldw + gcc]);
      return p ? v : z4;
    };

    uint4 ra0 = fetchA(0, ar0, ac0);
    uint4 ra1 = fetchA(0, ar1, ac1);
    uint4 rb  = fetchB(0, bn0, bc0);

    for (int k0 = 0; k0 < K; k0 += BK) {
      *reinterpret_cast<uint4*>(&As[0][ar0 * APAD + ac0]) = ra0;
      *reinterpret_cast<uint4*>(&As[0][ar1 * APAD + ac1]) = ra1;
      *reinterpret_cast<uint4*>(&Bs[0][bn0 * APAD + bc0]) = rb;
      __syncthreads();
      if (k0 + BK < K) {
        ra0 = fetchA(k0 + BK, ar0, ac0);
        ra1 = fetchA(k0 + BK, ar1, ac1);
        rb  = fetchB(k0 + BK, bn0, bc0);
      }
      do_wmma(0);
      __syncthreads();
    }
  }

  // Epilogue: C/D layout — element r of v8f: (m = half*8 + r, n = l16)
#pragma unroll
  for (int i = 0; i < 2; ++i)
#pragma unroll
    for (int j = 0; j < 2; ++j)
#pragma unroll
      for (int r = 0; r < 8; ++r) {
        int gr = row0 + wm * 32 + i * 16 + half * 8 + r;
        int gc = col0 + wn * 32 + j * 16 + l16;
        if (gr < M && gc < N) {
          float v = acc[i][j][r];
          if (bias) v += bias[gc];
          if (act == 1) v = (v > 20.f) ? v : __logf(1.f + __expf(v));   // softplus
          if (beta) v += C[(size_t)gr * ldc + gc];
          C[(size_t)gr * ldc + gc] = v;
          if (Cbf) Cbf[(size_t)gr * ldc + gc] = f32_to_bf16(v);
        }
      }
}

// ---------------------------------------------------------------------------
// Depthwise causal conv1d (k=4) + SiLU over the xc half of in_proj output
// ---------------------------------------------------------------------------
__global__ void conv_silu_kernel(const float* __restrict__ xz, const float* __restrict__ cw,
                                 const float* __restrict__ cb, float* __restrict__ xc,
                                 unsigned short* __restrict__ xcbf) {
  int idx = blockIdx.x * blockDim.x + threadIdx.x;
  if (idx >= M_ROWS * D_INNER) return;
  int d   = idx % D_INNER;
  int row = idx / D_INNER;
  int l = row % SEQ_L, b = row / SEQ_L;
  float s = cb[d];
#pragma unroll
  for (int j = 0; j < D_CONV; ++j) {
    int ll = l - (D_CONV - 1) + j;
    if (ll >= 0)
      s += xz[(size_t)(b * SEQ_L + ll) * (2 * D_INNER) + d] * cw[d * D_CONV + j];
  }
  s = s * sigmoid_f(s);
  xc[idx]   = s;
  xcbf[idx] = f32_to_bf16(s);
}

// ---------------------------------------------------------------------------
// Selective scan: one lane per (d, n) state; 16-lane shfl reduction for y.
//   grid = (D_INNER/16, B); block = 256 (16 d-channels x 16 states)
// ---------------------------------------------------------------------------
__global__ __launch_bounds__(256)
void scan_kernel(const float* __restrict__ delta, const float* __restrict__ u,
                 const float* __restrict__ dbl, const float* __restrict__ A_log,
                 const float* __restrict__ Dp, float* __restrict__ y) {
  const int tid  = threadIdx.x;
  const int n    = tid & 15;
  const int d    = blockIdx.x * 16 + (tid >> 4);
  const int b    = blockIdx.y;
  const float a  = -__expf(A_log[d * D_STATE + n]);
  const float Dd = Dp[d];
  float h = 0.f;
  const size_t base = (size_t)b * SEQ_L;
  for (int l = 0; l < SEQ_L; ++l) {
    const size_t r = base + l;
    float dl = delta[r * D_INNER + d];
    float uu = u[r * D_INNER + d];
    float bn = dbl[r * DBL_W + DT_RANK + n];
    float cn = dbl[r * DBL_W + DT_RANK + D_STATE + n];
    h = __expf(dl * a) * h + (dl * uu) * bn;
    float ys = h * cn;
    ys += __shfl_xor(ys, 8, 16);
    ys += __shfl_xor(ys, 4, 16);
    ys += __shfl_xor(ys, 2, 16);
    ys += __shfl_xor(ys, 1, 16);
    if (n == 0) y[r * D_INNER + d] = ys + uu * Dd;
  }
}

// ---------------------------------------------------------------------------
// Gate: yg = y * silu(res), emitted as bf16 for the out_proj GEMM
// ---------------------------------------------------------------------------
__global__ void gate_kernel(const float* __restrict__ y, const float* __restrict__ xz,
                            unsigned short* __restrict__ ygbf) {
  int idx = blockIdx.x * blockDim.x + threadIdx.x;
  if (idx >= M_ROWS * D_INNER) return;
  int d   = idx % D_INNER;
  int row = idx / D_INNER;
  float res = xz[(size_t)row * (2 * D_INNER) + D_INNER + d];
  ygbf[idx] = f32_to_bf16(y[idx] * (res * sigmoid_f(res)));
}

// ---------------------------------------------------------------------------
// RMS per row
// ---------------------------------------------------------------------------
__global__ __launch_bounds__(128)
void rms_kernel(const float* __restrict__ t, float* __restrict__ rms) {
  __shared__ float red[128];
  int row = blockIdx.x, tid = threadIdx.x;
  float s = 0.f;
  for (int j = tid; j < D_MODEL; j += 128) {
    float v = t[(size_t)row * D_MODEL + j];
    s += v * v;
  }
  red[tid] = s;
  __syncthreads();
  for (int o = 64; o > 0; o >>= 1) {
    if (tid < o) red[tid] += red[tid + o];
    __syncthreads();
  }
  if (tid == 0) rms[row] = sqrtf(red[0] / D_MODEL + 1e-5f);
}

// ---------------------------------------------------------------------------
// Mean-pool of (t / rms) * norm_w  ->  pooled[B, D_MODEL]
// ---------------------------------------------------------------------------
__global__ void pool_kernel(const float* __restrict__ t, const float* __restrict__ rms,
                            const float* __restrict__ nw, float* __restrict__ pooled) {
  int d = blockIdx.x * 128 + threadIdx.x;
  int b = blockIdx.y;
  if (d >= D_MODEL) return;
  float s = 0.f;
  for (int l = 0; l < SEQ_L; ++l) {
    size_t r = (size_t)b * SEQ_L + l;
    s += t[r * D_MODEL + d] / rms[r];
  }
  pooled[b * D_MODEL + d] = s * nw[d] * (1.f / SEQ_L);
}

// ---------------------------------------------------------------------------
// Classifier head (tiny: 4x1000x384)
// ---------------------------------------------------------------------------
__global__ void head_kernel(const float* __restrict__ pooled, const float* __restrict__ hw,
                            const float* __restrict__ hb, float* __restrict__ out) {
  int idx = blockIdx.x * blockDim.x + threadIdx.x;
  if (idx >= B_SZ * NUM_CLASSES) return;
  int n = idx % NUM_CLASSES, b = idx / NUM_CLASSES;
  float s = hb[n];
  for (int d = 0; d < D_MODEL; ++d) s += pooled[b * D_MODEL + d] * hw[n * D_MODEL + d];
  out[idx] = s;
}

// ---------------------------------------------------------------------------
// Host orchestration
// ---------------------------------------------------------------------------
static inline size_t ws_take(size_t& off, size_t bytes) {
  size_t o = off;
  off = (off + bytes + 255) & ~(size_t)255;
  return o;
}

extern "C" void kernel_launch(void* const* d_in, const int* in_sizes, int n_in,
                              void* d_out, int out_size, void* d_ws, size_t ws_size,
                              hipStream_t stream) {
  (void)in_sizes; (void)n_in; (void)out_size; (void)ws_size;

  const float* x         = (const float*)d_in[0];
  const float* patch_w   = (const float*)d_in[1];
  const float* patch_b   = (const float*)d_in[2];
  const float* in_proj_w = (const float*)d_in[3];
  const float* conv_w    = (const float*)d_in[4];
  const float* conv_b    = (const float*)d_in[5];
  const float* x_proj_w  = (const float*)d_in[6];
  const float* dt_proj_w = (const float*)d_in[7];
  const float* dt_proj_b = (const float*)d_in[8];
  const float* A_log     = (const float*)d_in[9];
  const float* Dp        = (const float*)d_in[10];
  const float* out_proj_w= (const float*)d_in[11];
  const float* norm_w    = (const float*)d_in[12];
  const float* head_w    = (const float*)d_in[13];
  const float* head_b    = (const float*)d_in[14];
  float* out = (float*)d_out;

  char* ws = (char*)d_ws;
  size_t off = 0;
  float*          t_f32   = (float*)(ws + ws_take(off, (size_t)M_ROWS * D_MODEL * 4));
  unsigned short* t_bf    = (unsigned short*)(ws + ws_take(off, (size_t)M_ROWS * D_MODEL * 2));
  float*          xz      = (float*)(ws + ws_take(off, (size_t)M_ROWS * 2 * D_INNER * 4));
  float*          xconv   = (float*)(ws + ws_take(off, (size_t)M_ROWS * D_INNER * 4));
  unsigned short* xconv_bf= (unsigned short*)(ws + ws_take(off, (size_t)M_ROWS * D_INNER * 2));
  float*          dbl     = (float*)(ws + ws_take(off, (size_t)M_ROWS * DBL_W * 4));
  unsigned short* dbl_bf  = (unsigned short*)(ws + ws_take(off, (size_t)M_ROWS * DBL_W * 2));
  float*          delta   = (float*)(ws + ws_take(off, (size_t)M_ROWS * D_INNER * 4));
  float*          ybuf    = (float*)(ws + ws_take(off, (size_t)M_ROWS * D_INNER * 4));
  unsigned short* yg_bf   = (unsigned short*)(ws + ws_take(off, (size_t)M_ROWS * D_INNER * 2));
  float*          rms     = (float*)(ws + ws_take(off, (size_t)M_ROWS * 4));
  float*          pooled  = (float*)(ws + ws_take(off, (size_t)B_SZ * D_MODEL * 4));
  unsigned short* w_in_bf = (unsigned short*)(ws + ws_take(off, (size_t)N_LAYERS * 2 * D_INNER * D_MODEL * 2));
  unsigned short* w_x_bf  = (unsigned short*)(ws + ws_take(off, (size_t)N_LAYERS * DBL_W * D_INNER * 2));
  unsigned short* w_dt_bf = (unsigned short*)(ws + ws_take(off, (size_t)N_LAYERS * D_INNER * DT_RANK * 2));
  unsigned short* w_out_bf= (unsigned short*)(ws + ws_take(off, (size_t)N_LAYERS * D_MODEL * D_INNER * 2));

  const int TPB = 256;
  auto cdiv = [](long long a, long long b) { return (int)((a + b - 1) / b); };

  // Weight conversion (f32 -> bf16), all layers at once
  {
    int n;
    n = N_LAYERS * 2 * D_INNER * D_MODEL;
    f2bf_kernel<<<cdiv(n, TPB), TPB, 0, stream>>>(in_proj_w, w_in_bf, n);
    n = N_LAYERS * DBL_W * D_INNER;
    f2bf_kernel<<<cdiv(n, TPB), TPB, 0, stream>>>(x_proj_w, w_x_bf, n);
    n = N_LAYERS * D_INNER * DT_RANK;
    f2bf_kernel<<<cdiv(n, TPB), TPB, 0, stream>>>(dt_proj_w, w_dt_bf, n);
    n = N_LAYERS * D_MODEL * D_INNER;
    f2bf_kernel<<<cdiv(n, TPB), TPB, 0, stream>>>(out_proj_w, w_out_bf, n);
  }

  // Patch embedding
  patch_embed_kernel<<<cdiv((long long)M_ROWS * D_MODEL, TPB), TPB, 0, stream>>>(
      x, patch_w, patch_b, t_f32, t_bf);

  for (int layer = 0; layer < N_LAYERS; ++layer) {
    const unsigned short* wi = w_in_bf  + (size_t)layer * 2 * D_INNER * D_MODEL;
    const unsigned short* wx = w_x_bf   + (size_t)layer * DBL_W * D_INNER;
    const unsigned short* wd = w_dt_bf  + (size_t)layer * D_INNER * DT_RANK;
    const unsigned short* wo = w_out_bf + (size_t)layer * D_MODEL * D_INNER;
    const float* cw  = conv_w    + (size_t)layer * D_INNER * D_CONV;
    const float* cb  = conv_b    + (size_t)layer * D_INNER;
    const float* dtb = dt_proj_b + (size_t)layer * D_INNER;
    const float* Al  = A_log     + (size_t)layer * D_INNER * D_STATE;
    const float* Dl  = Dp        + (size_t)layer * D_INNER;

    // in_proj: [M,384] x [1536,384]^T -> xz [M,1536]   (async fast path)
    {
      dim3 g(cdiv(2 * D_INNER, BN), cdiv(M_ROWS, BM));
      gemm_bf16_kernel<<<g, TPB, 0, stream>>>(t_bf, wi, nullptr, xz, nullptr,
                                              M_ROWS, 2 * D_INNER, D_MODEL,
                                              D_MODEL, D_MODEL, 2 * D_INNER, 0, 0);
    }
    // depthwise causal conv + silu
    conv_silu_kernel<<<cdiv((long long)M_ROWS * D_INNER, TPB), TPB, 0, stream>>>(
        xz, cw, cb, xconv, xconv_bf);
    // x_proj: [M,768] x [56,768]^T -> dbl [M,56] (+bf16)  (edge path, N=56)
    {
      dim3 g(cdiv(DBL_W, BN), cdiv(M_ROWS, BM));
      gemm_bf16_kernel<<<g, TPB, 0, stream>>>(xconv_bf, wx, nullptr, dbl, dbl_bf,
                                              M_ROWS, DBL_W, D_INNER,
                                              D_INNER, D_INNER, DBL_W, 0, 0);
    }
    // dt_proj + bias + softplus: [M,24] x [768,24]^T -> delta [M,768] (edge, K=24)
    {
      dim3 g(cdiv(D_INNER, BN), cdiv(M_ROWS, BM));
      gemm_bf16_kernel<<<g, TPB, 0, stream>>>(dbl_bf, wd, dtb, delta, nullptr,
                                              M_ROWS, D_INNER, DT_RANK,
                                              DBL_W, DT_RANK, D_INNER, 1, 0);
    }
    // selective scan
    {
      dim3 g(D_INNER / 16, B_SZ);
      scan_kernel<<<g, TPB, 0, stream>>>(delta, xconv, dbl, Al, Dl, ybuf);
    }
    // gate
    gate_kernel<<<cdiv((long long)M_ROWS * D_INNER, TPB), TPB, 0, stream>>>(ybuf, xz, yg_bf);
    // out_proj with residual add (beta=1), refresh t_bf  (async fast path)
    {
      dim3 g(cdiv(D_MODEL, BN), cdiv(M_ROWS, BM));
      gemm_bf16_kernel<<<g, TPB, 0, stream>>>(yg_bf, wo, nullptr, t_f32, t_bf,
                                              M_ROWS, D_MODEL, D_INNER,
                                              D_INNER, D_INNER, D_MODEL, 0, 1);
    }
  }

  // RMS norm + mean pool + head
  rms_kernel<<<M_ROWS, 128, 0, stream>>>(t_f32, rms);
  {
    dim3 g(cdiv(D_MODEL, 128), B_SZ);
    pool_kernel<<<g, 128, 0, stream>>>(t_f32, rms, norm_w, pooled);
  }
  head_kernel<<<cdiv(B_SZ * NUM_CLASSES, TPB), TPB, 0, stream>>>(pooled, head_w, head_b, out);
}
